// UNSATMinimizer_47459388621022
// MI455X (gfx1250) — matrix operand order repaired and checked by reading, not compiled, not run on hardware
//
#include <hip/hip_runtime.h>

namespace {

constexpr int kNV = 50000;
constexpr int kNC = 210000;
constexpr int kNE = 630000;
constexpr int kNG = 32;
constexpr int kRounds = 16;
constexpr float kEps = 1e-6f;

typedef __attribute__((ext_vector_type(16))) _Float16 v16h;
typedef __attribute__((ext_vector_type(8)))  float    v8f;

// ---------------------------------------------------------------- utilities
__global__ void fill_kernel(float* __restrict__ p, long n, float v) {
  long i = (long)blockIdx.x * blockDim.x + threadIdx.x;
  if (i < n) p[i] = v;
}

// pack weight [K,N] row-major f32 -> Bt [N][Kpad] f16 (transposed, zero-padded)
__global__ void pack_w_kernel(const float* __restrict__ w, _Float16* __restrict__ bt,
                              int K, int N, int Kpad) {
  int i = blockIdx.x * blockDim.x + threadIdx.x;
  if (i >= N * Kpad) return;
  const int n = i / Kpad;
  const int k = i - n * Kpad;
  bt[i] = (k < K) ? (_Float16)w[(size_t)k * N + n] : (_Float16)0.0f;
}

// noise_pad[v][0:4] = noise_t[v][0:4], cols 4..31 = 0
__global__ void noise_pad_kernel(const float* __restrict__ noise_t,
                                 float* __restrict__ pad) {
  int i = blockIdx.x * blockDim.x + threadIdx.x;
  if (i >= kNV * 32) return;
  const int j = i & 31;
  const int v = i >> 5;
  pad[i] = (j < 4) ? noise_t[v * 4 + j] : 0.0f;
}

__global__ void degree_kernel(const int* __restrict__ lit_idx, float* __restrict__ deg) {
  int e = blockIdx.x * blockDim.x + threadIdx.x;
  if (e < kNE) atomicAdd(&deg[lit_idx[e]], 1.0f);
}

__global__ void degw_kernel(const float* __restrict__ deg, float* __restrict__ degw,
                            float* __restrict__ vdw) {
  int i = blockIdx.x * blockDim.x + threadIdx.x;
  if (i < 2 * kNV) {
    degw[i] = rsqrtf(fmaxf(deg[i], 1.0f));
    if (i < kNV) vdw[i] = 4.0f * rsqrtf(fmaxf(deg[i] + deg[kNV + i], 1.0f));
  }
}

__global__ void count_kernel(const int* __restrict__ gid, int n, float* __restrict__ cnt) {
  int i = blockIdx.x * blockDim.x + threadIdx.x;
  if (i < n) atomicAdd(&cnt[gid[i]], 1.0f);
}

// ---------------------------------------------------------- WMMA tiled GEMM
// C[M,N] = concat(A0[:,0:S0*32], scale1*A1[:,0:S1*32]) @ B + bias, opt leaky(0.2)
// A0 lda = S0*32, A1 lda = S1*32 (row-major, contiguous).
// Bt: f16, transposed + padded: Bt[n][k], stride Kpad = (S0+S1)*32, 32B aligned.
// One wave computes a 16x32 tile (two accumulators share the A fetch).
// grid = (M/16, N/32), block = 32. A/B lanes hold contiguous k = 16h + s (same order for both); wave32.
template <int S0, int S1, bool LEAKY>
__global__ void __launch_bounds__(32)
gemm_wmma(const float* __restrict__ A0, const float* __restrict__ A1, float scale1,
          const _Float16* __restrict__ Bt, const float* __restrict__ bias,
          float* __restrict__ C, int N) {
  constexpr int NS = S0 + S1;
  constexpr int KP = NS * 32;
  const int lane = threadIdx.x & 31;
  const int hlf = lane >> 4;   // which K sub-range this lane holds
  const int l16 = lane & 15;
  const int mBase = blockIdx.x << 4;
  const int row = mBase + l16;           // A row held by this lane
  const int col0 = (blockIdx.y << 5) + l16;  // B/C columns held by this lane
  const int col1 = col0 + 16;

  const float* a0row = A0 + (size_t)row * (S0 * 32);
  const float* a1row = (S1 > 0) ? (A1 + (size_t)row * (S1 * 32)) : (const float*)0;
  const _Float16* bt0 = Bt + (size_t)col0 * KP + 16 * hlf;
  const _Float16* bt1 = Bt + (size_t)col1 * KP + 16 * hlf;

  v8f acc0 = {}, acc1 = {};
#pragma unroll
  for (int s = 0; s < NS; ++s) {
    // A lane data: 16 contiguous floats at k = s*32+16h
    const float* srow = (s < S0) ? (a0row + s * 32) : (a1row + (s - S0) * 32);
    float4 p0 = *(const float4*)(srow + 16 * hlf);
    float4 p1 = *(const float4*)(srow + 16 * hlf + 4);
    float4 p2 = *(const float4*)(srow + 16 * hlf + 8);
    float4 p3 = *(const float4*)(srow + 16 * hlf + 12);
    if (s >= S0) {  // folded per unrolled iteration
      p0.x *= scale1; p0.y *= scale1; p0.z *= scale1; p0.w *= scale1;
      p1.x *= scale1; p1.y *= scale1; p1.z *= scale1; p1.w *= scale1;
      p2.x *= scale1; p2.y *= scale1; p2.z *= scale1; p2.w *= scale1;
      p3.x *= scale1; p3.y *= scale1; p3.z *= scale1; p3.w *= scale1;
    }
    v16h a;
    a[0] = (_Float16)p0.x;  a[1] = (_Float16)p0.y;
    a[2] = (_Float16)p0.z;  a[3] = (_Float16)p0.w;
    a[4] = (_Float16)p1.x;  a[5] = (_Float16)p1.y;
    a[6] = (_Float16)p1.z;  a[7] = (_Float16)p1.w;
    a[8] = (_Float16)p2.x;  a[9] = (_Float16)p2.y;
    a[10] = (_Float16)p2.z; a[11] = (_Float16)p2.w;
    a[12] = (_Float16)p3.x; a[13] = (_Float16)p3.y;
    a[14] = (_Float16)p3.z; a[15] = (_Float16)p3.w;
    // B lane data: 16 contiguous f16 at Bt[col][s*32 + 16h]
    v16h b0 = *(const v16h*)(bt0 + s * 32);
    v16h b1 = *(const v16h*)(bt1 + s * 32);
    acc0 = __builtin_amdgcn_wmma_f32_16x16x32_f16(false, a, false, b0,
                                                  (short)0, acc0, false, false);
    acc1 = __builtin_amdgcn_wmma_f32_16x16x32_f16(false, a, false, b1,
                                                  (short)0, acc1, false, false);
  }
  const float bc0 = bias[col0];
  const float bc1 = bias[col1];
#pragma unroll
  for (int j = 0; j < 8; ++j) {
    const int r = mBase + j + 8 * hlf;   // C layout: VGPR j -> row j + 8*half
    float w0 = acc0[j] + bc0;
    float w1 = acc1[j] + bc1;
    if (LEAKY) {
      w0 = (w0 > 0.0f) ? w0 : 0.2f * w0;
      w1 = (w1 > 0.0f) ? w1 : 0.2f * w1;
    }
    C[(size_t)r * N + col0] = w0;
    C[(size_t)r * N + col1] = w1;
  }
}

// ----------------------------------------------------------- graph kernels
__device__ __forceinline__ float softplusf(float x) {
  return fmaxf(x, 0.0f) + log1pf(__expf(-fabsf(x)));
}

// cl[c,f] = exp(-sum_{j<3} softplus(+-query[lit])) ; clause c owns edges 3c..3c+2
__global__ void cl_kernel(const int* __restrict__ lit_idx,
                          const float* __restrict__ query, float* __restrict__ cl) {
  size_t i = (size_t)blockIdx.x * blockDim.x + threadIdx.x;
  if (i >= (size_t)kNC * 64) return;
  const int f = (int)(i & 63);
  const int c = (int)(i >> 6);
  float s = 0.0f;
#pragma unroll
  for (int j = 0; j < 3; ++j) {
    const int l = lit_idx[3 * c + j];
    const float q = (l < kNV) ? query[(size_t)l * 64 + f]
                              : -query[(size_t)(l - kNV) * 64 + f];
    s += softplusf(q);
  }
  cl[i] = __expf(-s);
}

// glit[l,f] -= cl[clause(e),f] for each edge e with lit_idx[e]==l
__global__ void scatter_glit_kernel(const int* __restrict__ lit_idx,
                                    const int* __restrict__ clause_idx,
                                    const float* __restrict__ cl,
                                    float* __restrict__ glit) {
  size_t i = (size_t)blockIdx.x * blockDim.x + threadIdx.x;
  if (i >= (size_t)kNE * 64) return;
  const int f = (int)(i & 63);
  const int e = (int)(i >> 6);
  const int l = lit_idx[e];
  const int c = clause_idx[e];
  atomicAdd(&glit[(size_t)l * 64 + f], -cl[(size_t)c * 64 + f]);
}

// unit[:,0:64] = vgrad * vdw ; unit[:,64:128] = variables
__global__ void vgrad_unit_kernel(const float* __restrict__ query,
                                  const float* __restrict__ glit,
                                  const float* __restrict__ vdw,
                                  const float* __restrict__ variables,
                                  float* __restrict__ unit) {
  size_t i = (size_t)blockIdx.x * blockDim.x + threadIdx.x;
  if (i >= (size_t)kNV * 64) return;
  const int f = (int)(i & 63);
  const int v = (int)(i >> 6);
  const float q  = query[i];
  const float sp = 1.0f / (1.0f + __expf(-q));   // sigmoid(q)
  const float sn = 1.0f / (1.0f + __expf(q));    // sigmoid(-q)
  const float g  = glit[(size_t)v * 64 + f] * sp
                 - glit[(size_t)(kNV + v) * 64 + f] * sn;
  unit[(size_t)v * 256 + f]      = g * vdw[v];
  unit[(size_t)v * 256 + 64 + f] = variables[i];
}

__global__ void zero_unit_loss_kernel(float* __restrict__ unit) {
  size_t i = (size_t)blockIdx.x * blockDim.x + threadIdx.x;
  if (i >= (size_t)kNV * 128) return;
  const int f = (int)(i & 127);
  const int v = (int)(i >> 7);
  unit[(size_t)v * 256 + 128 + f] = 0.0f;
}

// scatter variables_loss_all = cdata[:,0:64] into unit cols 128..255
__global__ void scatter_vloss_kernel(const int* __restrict__ lit_idx,
                                     const int* __restrict__ clause_idx,
                                     const float* __restrict__ cdata,
                                     float* __restrict__ unit) {
  size_t i = (size_t)blockIdx.x * blockDim.x + threadIdx.x;
  if (i >= (size_t)kNE * 64) return;
  const int f = (int)(i & 63);
  const int e = (int)(i >> 6);
  const int l = lit_idx[e];
  const int c = clause_idx[e];
  const int rowv = (l < kNV) ? l : (l - kNV);
  const int colb = (l < kNV) ? 128 : 192;
  atomicAdd(&unit[(size_t)rowv * 256 + colb + f], cdata[(size_t)c * 128 + f]);
}

__global__ void scale_vloss_kernel(float* __restrict__ unit,
                                   const float* __restrict__ degw) {
  size_t i = (size_t)blockIdx.x * blockDim.x + threadIdx.x;
  if (i >= (size_t)kNV * 128) return;
  const int f = (int)(i & 127);
  const int v = (int)(i >> 7);
  const int l = (f < 64) ? v : (kNV + v);
  unit[(size_t)v * 256 + 128 + f] *= degw[l];
}

// ------------------------------------------------------------- pair norm
__global__ void pn_reduce_kernel(const float* __restrict__ x, int ldx, int n,
                                 const int* __restrict__ seg,
                                 float* __restrict__ sum, float* __restrict__ sqs) {
  const size_t gt = (size_t)blockIdx.x * blockDim.x + threadIdx.x;
  const int f = (int)(gt & 63);
  const int chunk = (int)(gt >> 6);
  int r0 = chunk * 64;
  if (r0 >= n) return;
  int r1 = min(r0 + 64, n);
  int cur = seg[r0];
  float s = 0.0f, q = 0.0f;
  for (int r = r0; r < r1; ++r) {
    const int g = seg[r];
    if (g != cur) {
      atomicAdd(&sum[cur * 64 + f], s);
      atomicAdd(&sqs[cur * 64 + f], q);
      s = 0.0f; q = 0.0f; cur = g;
    }
    const float v = x[(size_t)r * ldx + f];
    s += v; q += v * v;
  }
  atomicAdd(&sum[cur * 64 + f], s);
  atomicAdd(&sqs[cur * 64 + f], q);
}

// out = pair_norm(x)*0.25 + 0.1*out
__global__ void pn_apply_kernel(const float* __restrict__ x, int ldx, int n,
                                const int* __restrict__ seg,
                                const float* __restrict__ sum,
                                const float* __restrict__ sqs,
                                const float* __restrict__ cnt,
                                float* __restrict__ out) {
  size_t i = (size_t)blockIdx.x * blockDim.x + threadIdx.x;
  if (i >= (size_t)n * 64) return;
  const int f = (int)(i & 63);
  const int r = (int)(i >> 6);
  const int g = seg[r];
  const float c    = cnt[g];
  const float mean = sum[g * 64 + f] / c;
  float var = sqs[g * 64 + f] / c - mean * mean;
  var = fmaxf(var, 0.0f);
  const float xv = x[(size_t)r * ldx + f];
  const float nv = (xv - mean) * rsqrtf(var + kEps);
  out[(size_t)r * 64 + f] = nv * 0.25f + 0.1f * out[(size_t)r * 64 + f];
}

// ------------------------------------------------------------- output head
__global__ void out_logits_kernel(const float* __restrict__ h,
                                  const float* __restrict__ oW1,
                                  const float* __restrict__ ob1,
                                  float* __restrict__ out) {
  const int c = blockIdx.x * blockDim.x + threadIdx.x;
  if (c >= kNC) return;
  float s = ob1[0];
#pragma unroll 8
  for (int f = 0; f < 64; ++f) s += h[(size_t)c * 64 + f] * oW1[f];
  out[c]       = 1.0f / (1.0f + __expf(-s));   // sigmoid
  out[kNC + c] = softplusf(s);                 // softplus
}

} // namespace

// ---------------------------------------------------------------- launcher
extern "C" void kernel_launch(void* const* d_in, const int* in_sizes, int n_in,
                              void* d_out, int out_size, void* d_ws, size_t ws_size,
                              hipStream_t stream) {
  (void)in_sizes; (void)n_in; (void)out_size; (void)ws_size;

  const int*   lit_idx    = (const int*)d_in[0];
  const int*   clause_idx = (const int*)d_in[1];
  const int*   var_gid    = (const int*)d_in[2];
  const int*   clause_gid = (const int*)d_in[3];
  const float* noise      = (const float*)d_in[4];
  const float* qW0 = (const float*)d_in[5];  const float* qb0 = (const float*)d_in[6];
  const float* qW1 = (const float*)d_in[7];  const float* qb1 = (const float*)d_in[8];
  const float* cW0 = (const float*)d_in[9];  const float* cb0 = (const float*)d_in[10];
  const float* cW1 = (const float*)d_in[11]; const float* cb1 = (const float*)d_in[12];
  const float* uW0 = (const float*)d_in[13]; const float* ub0 = (const float*)d_in[14];
  const float* uW1 = (const float*)d_in[15]; const float* ub1 = (const float*)d_in[16];
  const float* uW2 = (const float*)d_in[17]; const float* ub2 = (const float*)d_in[18];
  const float* oW0 = (const float*)d_in[19]; const float* ob0 = (const float*)d_in[20];
  const float* oW1 = (const float*)d_in[21]; const float* ob1 = (const float*)d_in[22];

  // -------- workspace carve-up (floats)
  float* W = (float*)d_ws;
  size_t o = 0;
  auto alloc = [&](size_t n) { float* p = W + o; o += n; return p; };
  float* variables = alloc((size_t)kNV * 64);
  float* clauses   = alloc((size_t)kNC * 64);
  float* query     = alloc((size_t)kNV * 64);
  float* cl        = alloc((size_t)kNC * 64);
  float* glit      = alloc((size_t)2 * kNV * 64);
  float* chid      = alloc((size_t)kNC * 128);
  float* cdata     = alloc((size_t)kNC * 128);
  float* unit      = alloc((size_t)kNV * 256);
  float* vh1       = alloc((size_t)kNV * 128);
  float* vh2       = alloc((size_t)kNV * 128);
  float* vout      = alloc((size_t)kNV * 64);
  float* npad      = alloc((size_t)kNV * 32);
  float* pn_sum    = alloc(kNG * 64);
  float* pn_sqs    = alloc(kNG * 64);
  float* cntv      = alloc(kNG);
  float* cntc      = alloc(kNG);
  float* deg       = alloc(2 * kNV);
  float* degw      = alloc(2 * kNV);
  float* vdw       = alloc(kNV);
  o = (o + 15) & ~(size_t)15;  // 64B-align the f16 region (b128 loads)
  _Float16* H = (_Float16*)(W + o);
  size_t ho = 0;
  auto allocH = [&](size_t n) { _Float16* p = H + ho; ho += n; return p; };
  _Float16* qW0h = allocH((size_t)64 * 96);    // [N=64][Kpad=96]
  _Float16* qW1h = allocH((size_t)64 * 64);    // [64][64]
  _Float16* cW0h = allocH((size_t)128 * 128);  // [128][128]
  _Float16* cW1h = allocH((size_t)128 * 128);
  _Float16* uW0h = allocH((size_t)128 * 256);  // [128][256]
  _Float16* uW1h = allocH((size_t)128 * 128);
  _Float16* uW2h = allocH((size_t)64 * 128);   // [64][128]
  _Float16* oW0h = allocH((size_t)64 * 64);

  auto gdiv = [](long n, long b) { return (unsigned)((n + b - 1) / b); };
  auto fill = [&](float* p, long n, float v) {
    fill_kernel<<<gdiv(n, 256), 256, 0, stream>>>(p, n, v);
  };
  auto packw = [&](const float* w, _Float16* bt, int K, int N, int Kpad) {
    pack_w_kernel<<<gdiv((long)N * Kpad, 256), 256, 0, stream>>>(w, bt, K, N, Kpad);
  };

  // -------- one-time (per call) setup
  packw(qW0, qW0h, 68, 64, 96);
  packw(qW1, qW1h, 64, 64, 64);
  packw(cW0, cW0h, 128, 128, 128);
  packw(cW1, cW1h, 128, 128, 128);
  packw(uW0, uW0h, 256, 128, 256);
  packw(uW1, uW1h, 128, 128, 128);
  packw(uW2, uW2h, 128, 64, 128);
  packw(oW0, oW0h, 64, 64, 64);

  fill(variables, (long)kNV * 64, 1.0f);
  fill(clauses, (long)kNC * 64, 1.0f);
  fill(deg, 2 * kNV, 0.0f);
  fill(cntv, kNG, 0.0f);
  fill(cntc, kNG, 0.0f);
  degree_kernel<<<gdiv(kNE, 256), 256, 0, stream>>>(lit_idx, deg);
  degw_kernel<<<gdiv(2 * kNV, 256), 256, 0, stream>>>(deg, degw, vdw);
  count_kernel<<<gdiv(kNV, 256), 256, 0, stream>>>(var_gid, kNV, cntv);
  count_kernel<<<gdiv(kNC, 256), 256, 0, stream>>>(clause_gid, kNC, cntc);

  const long nNVf = (long)kNV * 64, nNCf = (long)kNC * 64, nNEf = (long)kNE * 64;
  const dim3 gV64(kNV / 16, 64 / 32), gV128(kNV / 16, 128 / 32);
  const dim3 gC64(kNC / 16, 64 / 32), gC128(kNC / 16, 128 / 32);

  for (int t = 0; t < kRounds; ++t) {
    const float* noise_t = noise + (size_t)t * kNV * 4;
    noise_pad_kernel<<<gdiv((long)kNV * 32, 256), 256, 0, stream>>>(noise_t, npad);
    // query = mlp2([variables | noise_t])
    gemm_wmma<2, 1, true><<<gV64, 32, 0, stream>>>(variables, npad, 1.0f,
                                                   qW0h, qb0, vh1, 64);
    gemm_wmma<2, 0, false><<<gV64, 32, 0, stream>>>(vh1, nullptr, 0.0f,
                                                    qW1h, qb1, query, 64);
    // clause losses cl = exp(-cv)
    cl_kernel<<<gdiv(nNCf, 256), 256, 0, stream>>>(lit_idx, query, cl);
    // gradient scatter + vgrad into unit[:,0:128]
    fill(glit, (long)2 * kNV * 64, 0.0f);
    scatter_glit_kernel<<<gdiv(nNEf, 256), 256, 0, stream>>>(lit_idx, clause_idx, cl, glit);
    vgrad_unit_kernel<<<gdiv(nNVf, 256), 256, 0, stream>>>(query, glit, vdw, variables, unit);
    // clause MLP on [clauses | 4*cl]
    gemm_wmma<2, 2, true><<<gC128, 32, 0, stream>>>(clauses, cl, 4.0f,
                                                    cW0h, cb0, chid, 128);
    gemm_wmma<4, 0, false><<<gC128, 32, 0, stream>>>(chid, nullptr, 0.0f,
                                                     cW1h, cb1, cdata, 128);
    // variables_loss scatter into unit[:,128:256], then * degree_weight
    zero_unit_loss_kernel<<<gdiv((long)kNV * 128, 256), 256, 0, stream>>>(unit);
    scatter_vloss_kernel<<<gdiv(nNEf, 256), 256, 0, stream>>>(lit_idx, clause_idx, cdata, unit);
    scale_vloss_kernel<<<gdiv((long)kNV * 128, 256), 256, 0, stream>>>(unit, degw);
    // clause pair-norm on cdata[:,64:128] -> clauses
    fill(pn_sum, kNG * 64, 0.0f);
    fill(pn_sqs, kNG * 64, 0.0f);
    pn_reduce_kernel<<<gdiv((long)gdiv(kNC, 64) * 64, 256), 256, 0, stream>>>(
        cdata + 64, 128, kNC, clause_gid, pn_sum, pn_sqs);
    pn_apply_kernel<<<gdiv(nNCf, 256), 256, 0, stream>>>(
        cdata + 64, 128, kNC, clause_gid, pn_sum, pn_sqs, cntc, clauses);
    // variable update MLP3 on unit
    gemm_wmma<8, 0, true><<<gV128, 32, 0, stream>>>(unit, nullptr, 0.0f,
                                                    uW0h, ub0, vh1, 128);
    gemm_wmma<4, 0, true><<<gV128, 32, 0, stream>>>(vh1, nullptr, 0.0f,
                                                    uW1h, ub1, vh2, 128);
    gemm_wmma<4, 0, false><<<gV64, 32, 0, stream>>>(vh2, nullptr, 0.0f,
                                                    uW2h, ub2, vout, 64);
    // variable pair-norm -> variables
    fill(pn_sum, kNG * 64, 0.0f);
    fill(pn_sqs, kNG * 64, 0.0f);
    pn_reduce_kernel<<<gdiv((long)gdiv(kNV, 64) * 64, 256), 256, 0, stream>>>(
        vout, 64, kNV, var_gid, pn_sum, pn_sqs);
    pn_apply_kernel<<<gdiv(nNVf, 256), 256, 0, stream>>>(
        vout, 64, kNV, var_gid, pn_sum, pn_sqs, cntv, variables);
  }

  // output head (only last round's logits matter)
  gemm_wmma<2, 0, true><<<gC64, 32, 0, stream>>>(clauses, nullptr, 0.0f,
                                                 oW0h, ob0, chid, 64);
  out_logits_kernel<<<gdiv(kNC, 256), 256, 0, stream>>>(chid, oW1, ob1, (float*)d_out);
}
